// LanguageContinuousThoughtMachine_6940667150756
// MI455X (gfx1250) — matrix operand order repaired and checked by reading, not compiled, not run on hardware
//
#include <hip/hip_runtime.h>
#include <hip/hip_bf16.h>

// ---------------- problem constants (match reference) ----------------
#define B_  2
#define L_  512
#define Q_  512
#define E_  1024
#define W_  1024
#define D_  8
#define HN_ 32      // H_NLM
#define NH_ 16
#define HD_ 64
#define SA_ 512
#define SO_ 1024
#define V_  8192
#define T_  6
#define SH_ 2048    // SYN_H
#define BQ_ (B_*Q_) // == B_*L_ == 1024

typedef __attribute__((ext_vector_type(16))) __bf16 v16bf;
typedef __attribute__((ext_vector_type(8)))  float  v8f;

__device__ __forceinline__ v8f wmma_bf(v16bf a, v16bf b, v8f c) {
  // D = A(16x32 bf16) * B(32x16 bf16) + C(16x16 f32)
  return __builtin_amdgcn_wmma_f32_16x16x32_bf16(false, a, false, b, (short)0, c, false, false);
}

__device__ __forceinline__ float gelu_f(float x) {
  const float c = 0.7978845608028654f; // sqrt(2/pi)
  return 0.5f * x * (1.0f + tanhf(c * (x + 0.044715f * x * x * x)));
}

// ---------------- elementwise helpers ----------------
__global__ void convert_bf16_kernel(const float* __restrict__ src, __bf16* __restrict__ dst, long n) {
  long i = (long)blockIdx.x * blockDim.x + threadIdx.x;
  long st = (long)gridDim.x * blockDim.x;
  for (; i < n; i += st) dst[i] = (__bf16)src[i];
}

__global__ void gather_embed_kernel(const int* __restrict__ x, const float* __restrict__ emb,
                                    __bf16* __restrict__ dst) {
  long i = (long)blockIdx.x * blockDim.x + threadIdx.x;
  if (i >= (long)B_ * L_ * E_) return;
  int tok = x[i / E_];
  dst[i] = (__bf16)emb[(size_t)tok * E_ + (i % E_)];
}

__global__ void hist0_kernel(const float* __restrict__ z_init, float* __restrict__ h0) {
  long i = (long)blockIdx.x * blockDim.x + threadIdx.x;
  if (i >= (long)BQ_ * W_) return;
  h0[i] = z_init[i % W_];
}

// preact[:, E_ + w] = bf16(z[:, w])
__global__ void zcat_kernel(const float* __restrict__ z, __bf16* __restrict__ preact) {
  long i = (long)blockIdx.x * blockDim.x + threadIdx.x;
  if (i >= (long)BQ_ * W_) return;
  long bq = i / W_, w = i % W_;
  preact[bq * (E_ + W_) + E_ + w] = (__bf16)z[i];
}

// proj (B x S x E f32, e = h*HD+d) -> RoPE -> outh [b][h][s][d] bf16
__global__ void rope_kernel(const float* __restrict__ proj, __bf16* __restrict__ outh, int S) {
  int tid = blockIdx.x * blockDim.x + threadIdx.x;
  if (tid >= B_ * S * NH_ * (HD_ / 2)) return;
  int i  = tid & 31;
  int h  = (tid >> 5) & (NH_ - 1);
  int s  = (tid >> 9) % S;
  int b  = (tid >> 9) / S;
  float inv = powf(10000.f, -(float)i / 32.f);
  float ang = (float)s * inv;
  float sn, cs;
  __sincosf(ang, &sn, &cs);
  const float* pr = proj + ((size_t)(b * S + s)) * E_ + h * HD_;
  float x1 = pr[i], x2 = pr[i + 32];
  __bf16* o = outh + (((size_t)(b * NH_ + h)) * S + s) * HD_;
  o[i]      = (__bf16)(x1 * cs - x2 * sn);
  o[i + 32] = (__bf16)(x2 * cs + x1 * sn);
}

// vproj (B x L x E f32) -> vt [b][h][d][k] bf16 (transposed for WMMA B-fragments)
__global__ void vresh_kernel(const float* __restrict__ vproj, __bf16* __restrict__ vt) {
  long tid = (long)blockIdx.x * blockDim.x + threadIdx.x;
  if (tid >= (long)B_ * NH_ * HD_ * L_) return;
  int k = tid % L_;
  int d = (tid / L_) % HD_;
  int h = (tid / ((long)L_ * HD_)) % NH_;
  int b = tid / ((long)L_ * HD_ * NH_);
  vt[tid] = (__bf16)vproj[((size_t)(b * L_ + k)) * E_ + h * HD_ + d];
}

// sync[b,q,s] = sum_t zl*e*zr / sqrt(sum_t e),  e = exp(-(Tn-1-t)*decay[s])
__global__ void sync_kernel(const float* __restrict__ hist, int Tn,
                            const int* __restrict__ il, const int* __restrict__ ir,
                            const float* __restrict__ dc, int S, __bf16* __restrict__ out) {
  long tid = (long)blockIdx.x * blockDim.x + threadIdx.x;
  if (tid >= (long)BQ_ * S) return;
  int s = tid % S;
  long bq = tid / S;
  int a = il[s], b = ir[s];
  float d = dc[s];
  float num = 0.f, den = 0.f;
  for (int t = 0; t < Tn; ++t) {
    float e = __expf(-(float)(Tn - 1 - t) * d);
    const float* hs = hist + (size_t)t * BQ_ * W_ + (size_t)bq * W_;
    num += hs[a] * e * hs[b];
    den += e;
  }
  out[tid] = (__bf16)(num * rsqrtf(den));
}

// ---------------- generic bf16 WMMA GEMM (async-to-LDS, double buffered) ----------------
// C[M,N] = act(A[M,K] @ B[K,N] + bias); A ld=K, B ld=N row-major.
// Writes f32 (Cf, ldcf) and/or bf16 (Cb, ldcb). M,N mult of 64; K mult of 64.
__global__ __launch_bounds__(128)
void gemm_bf16_kernel(const __bf16* __restrict__ A, const __bf16* __restrict__ Bm,
                      const float* __restrict__ bias,
                      float* Cf, int ldcf, __bf16* Cb, int ldcb,
                      int M, int N, int K, int act_gelu) {
  __shared__ __bf16 As[2][64][72];   // [buf][m][k], row stride 144B (16B aligned)
  __shared__ __bf16 Bst[2][64][72];  // [buf][n][k] transposed
  const int tid  = threadIdx.x;
  const int lane = tid & 31;
  const int wid  = tid >> 5;
  const int wm   = (wid >> 1) * 32;
  const int wn   = (wid & 1) * 32;
  const int m0   = blockIdx.y * 64;
  const int n0   = blockIdx.x * 64;

  v8f acc[2][2];
  v8f zero = {};
#pragma unroll
  for (int i = 0; i < 2; ++i)
#pragma unroll
    for (int j = 0; j < 2; ++j) acc[i][j] = zero;

  const int arow = tid >> 1, acol = (tid & 1) * 32;  // A: 64B (32 bf16) per thread
  const int brow = tid >> 1, bseg = (tid & 1) * 32;  // B: 64B (32 cols) per thread
  const int lrow = lane & 15;
  const int hi   = (lane < 16) ? 0 : 1;

  // --- stage helpers ---
  // A tile (64x64 bf16 = 8KB): DMA straight into LDS, no VGPR round-trip.
  // INST_OFFSET applies to both LDS and memory address -> contiguous 64B copy.
  auto a_async = [&](int k0, int bb) {
    const __bf16* g = A + (size_t)(m0 + arow) * K + k0 + acol;
    unsigned l = (unsigned)(unsigned long long)&As[bb][arow][acol];
    asm volatile("global_load_async_to_lds_b128 %0, %1, off"            :: "v"(l), "v"(g) : "memory");
    asm volatile("global_load_async_to_lds_b128 %0, %1, off offset:16"  :: "v"(l), "v"(g) : "memory");
    asm volatile("global_load_async_to_lds_b128 %0, %1, off offset:32"  :: "v"(l), "v"(g) : "memory");
    asm volatile("global_load_async_to_lds_b128 %0, %1, off offset:48"  :: "v"(l), "v"(g) : "memory");
  };
  auto b_load = [&](int k0, uint4* r) {
    const uint4* src = (const uint4*)(Bm + (size_t)(k0 + brow) * N + n0 + bseg);
#pragma unroll
    for (int v = 0; v < 4; ++v) r[v] = src[v];
  };
  auto b_scatter = [&](int bb, const uint4* r) {
#pragma unroll
    for (int v = 0; v < 4; ++v) {
      const __bf16* t = (const __bf16*)&r[v];
#pragma unroll
      for (int i = 0; i < 8; ++i) Bst[bb][bseg + v * 8 + i][brow] = t[i];
    }
  };
  auto compute = [&](int bb) {
#pragma unroll
    for (int c = 0; c < 2; ++c) {  // two K=32 chunks per stage
      const int akb = c * 32 + (hi ? 8 : 0);   // A-frag K base (ISA 16x32 layout)
      const int bkb = c * 32 + (hi ? 8 : 0);   // B-frag K base (ISA 32x16 layout)
      v16bf af[2], bf2[2];
#pragma unroll
      for (int i = 0; i < 2; ++i) {
        const __bf16* ap = &As[bb][wm + i * 16 + lrow][0];
#pragma unroll
        for (int j = 0; j < 8; ++j) { af[i][j] = ap[akb + j]; af[i][j + 8] = ap[akb + 16 + j]; }
        const __bf16* bp = &Bst[bb][wn + i * 16 + lrow][0];
#pragma unroll
        for (int j = 0; j < 8; ++j) { bf2[i][j] = bp[bkb + j]; bf2[i][j + 8] = bp[bkb + 16 + j]; }
      }
#pragma unroll
      for (int i = 0; i < 2; ++i)
#pragma unroll
        for (int j = 0; j < 2; ++j) acc[i][j] = wmma_bf(af[i], bf2[j], acc[i][j]);
    }
  };

  // --- pipeline: prologue stage 0, then overlap stage s+1 loads with stage s WMMAs ---
  const int nst = K >> 6;
  uint4 br[4];
  a_async(0, 0);
  b_load(0, br);
  b_scatter(0, br);
  asm volatile("s_wait_asynccnt 0x0" ::: "memory");
  __syncthreads();

  int cur = 0;
  for (int s = 0; s < nst; ++s) {
    const int nxt = cur ^ 1;
    const bool more = (s + 1 < nst);
    if (more) {
      a_async((s + 1) << 6, nxt);   // ASYNCcnt path, overlaps with compute below
      b_load((s + 1) << 6, br);     // global latency hidden under WMMAs
      if (s + 2 < nst) __builtin_prefetch(Bm + (size_t)(((s + 2) << 6) + brow) * N + n0 + bseg, 0, 0);
    }
    compute(cur);                    // 8 x v_wmma_f32_16x16x32_bf16 per wave
    if (more) {
      b_scatter(nxt, br);
      asm volatile("s_wait_asynccnt 0x0" ::: "memory");
    }
    __syncthreads();
    cur = nxt;
  }

  // --- epilogue ---
#pragma unroll
  for (int i = 0; i < 2; ++i)
#pragma unroll
    for (int j = 0; j < 2; ++j) {
      const int col = n0 + wn + j * 16 + lrow;
      float bv = bias ? bias[col] : 0.f;
#pragma unroll
      for (int r = 0; r < 8; ++r) {
        const int row = m0 + wm + i * 16 + r + hi * 8;
        float v = acc[i][j][r] + bv;
        if (act_gelu) v = gelu_f(v);
        if (Cf) Cf[(size_t)row * ldcf + col] = v;
        if (Cb) Cb[(size_t)row * ldcb + col] = (__bf16)v;
      }
    }
}

// ---------------- fused causal attention (flash-style, WMMA) ----------------
// qh/kh: [b][h][s][d] bf16 (RoPEd, q pre-scaled here), vt: [b][h][d][k] bf16
// out attnc: [b][q][h*HD+d] bf16
__global__ __launch_bounds__(128)
void attention_kernel(const __bf16* __restrict__ qh, const __bf16* __restrict__ kh,
                      const __bf16* __restrict__ vt, __bf16* __restrict__ attnc) {
  __shared__ float sS[4][16][36];
  __shared__ float sAlpha[4][16];
  __shared__ float sLinv[4][16];
  const int tid = threadIdx.x, lane = tid & 31, wid = tid >> 5;
  int blk = blockIdx.x;
  const int qt = blk & 7;  blk >>= 3;
  const int h  = blk & 15; blk >>= 4;
  const int b  = blk;
  const int qbase = qt * 64 + wid * 16;
  const int lrow = lane & 15;
  const int hi = (lane < 16) ? 0 : 1;

  // Q fragments (scaled by 1/sqrt(HD))
  v16bf a0, a1;
  {
    const __bf16* qrow = qh + (((size_t)(b * NH_ + h)) * Q_ + qbase + lrow) * HD_;
    const int kb = hi ? 8 : 0;
#pragma unroll
    for (int j = 0; j < 8; ++j) {
      a0[j]     = (__bf16)((float)qrow[kb + j] * 0.125f);
      a0[j + 8] = (__bf16)((float)qrow[kb + 16 + j] * 0.125f);
      a1[j]     = (__bf16)((float)qrow[32 + kb + j] * 0.125f);
      a1[j + 8] = (__bf16)((float)qrow[32 + kb + 16 + j] * 0.125f);
    }
  }
  v8f zero = {};
  v8f o[4];
#pragma unroll
  for (int d = 0; d < 4; ++d) o[d] = zero;
  float mreg = -1e30f, lreg = 0.f;

  const __bf16* khh = kh + ((size_t)(b * NH_ + h)) * L_ * HD_;
  const __bf16* vth = vt + ((size_t)(b * NH_ + h)) * HD_ * L_;
  const int ntiles = (qbase + 15) / 32 + 1;  // wave-uniform causal bound

  for (int kt = 0; kt < ntiles; ++kt) {
    // scores: 16q x 32k via 2 col-blocks x 2 K-chunk WMMAs
#pragma unroll
    for (int cb = 0; cb < 2; ++cb) {
      const int key = kt * 32 + cb * 16 + lrow;
      const __bf16* krow = khh + (size_t)key * HD_;
      const int kb2 = hi ? 8 : 0;
      v16bf b0, b1;
#pragma unroll
      for (int j = 0; j < 8; ++j) { b0[j] = krow[kb2 + j]; b0[j + 8] = krow[kb2 + 16 + j]; b1[j] = krow[32 + kb2 + j]; b1[j + 8] = krow[32 + kb2 + 16 + j]; }
      v8f s = zero;
      s = wmma_bf(a0, b0, s);
      s = wmma_bf(a1, b1, s);
#pragma unroll
      for (int r = 0; r < 8; ++r) {
        int row = r + hi * 8;
        int qg = qbase + row;
        int kg = kt * 32 + cb * 16 + lrow;
        sS[wid][row][cb * 16 + lrow] = (kg <= qg) ? s[r] : -1e30f;
      }
    }
    asm volatile("" ::: "memory");  // per-wave LDS RAW: DS pipe is in-order within a wave
    if (lane < 16) {                // one row per lane 0..15
      float mold = mreg, mx = mold;
#pragma unroll
      for (int k = 0; k < 32; ++k) mx = fmaxf(mx, sS[wid][lane][k]);
      float alpha = __expf(mold - mx);
      float l = lreg * alpha;
#pragma unroll
      for (int k = 0; k < 32; ++k) {
        float p = __expf(sS[wid][lane][k] - mx);
        sS[wid][lane][k] = p;
        l += p;
      }
      sAlpha[wid][lane] = alpha;
      mreg = mx; lreg = l;
    }
    asm volatile("" ::: "memory");
    // rescale O accumulators by per-row alpha
#pragma unroll
    for (int r = 0; r < 8; ++r) {
      float a = sAlpha[wid][r + hi * 8];
#pragma unroll
      for (int d = 0; d < 4; ++d) o[d][r] *= a;
    }
    // P fragment from LDS
    v16bf p;
    {
      const float* prow = &sS[wid][lrow][0];
      const int kbp = hi ? 8 : 0;
#pragma unroll
      for (int j = 0; j < 8; ++j) { p[j] = (__bf16)prow[kbp + j]; p[j + 8] = (__bf16)prow[kbp + 16 + j]; }
    }
    // O += P(16x32) @ V(32x64)
#pragma unroll
    for (int d = 0; d < 4; ++d) {
      const __bf16* vcol = vth + (size_t)(d * 16 + lrow) * L_ + kt * 32 + (hi ? 8 : 0);
      v16bf vf;
#pragma unroll
      for (int j = 0; j < 8; ++j) { vf[j] = vcol[j]; vf[j + 8] = vcol[16 + j]; }
      o[d] = wmma_bf(p, vf, o[d]);
    }
  }
  if (lane < 16) sLinv[wid][lane] = 1.0f / lreg;
  asm volatile("" ::: "memory");
#pragma unroll
  for (int r = 0; r < 8; ++r) {
    int row = r + hi * 8;
    float sc = sLinv[wid][row];
    __bf16* orow = attnc + ((size_t)(b * Q_ + qbase + row)) * E_ + h * HD_;
#pragma unroll
    for (int d = 0; d < 4; ++d) orow[d * 16 + lrow] = (__bf16)(o[d][r] * sc);
  }
}

// ---------------- NLM (per-w tiny MLP over pah window) ----------------
// pah never materialized: slot d = (d < 7-t) ? pah_init[w][d+t+1] : pre_hist[d-7+t]
__global__ __launch_bounds__(256)
void nlm_kernel(const float* __restrict__ pre_hist, const float* __restrict__ pah_init,
                const float* __restrict__ w1, const float* __restrict__ b1,
                const float* __restrict__ w2, const float* __restrict__ b2,
                float* __restrict__ zout, int t) {
  __shared__ float w1s[D_ * HN_];
  __shared__ float b1s[HN_], w2s[HN_];
  const int w = blockIdx.y;
  const int tid = threadIdx.x;
  w1s[tid] = w1[(size_t)w * D_ * HN_ + tid];
  if (tid < HN_) { b1s[tid] = b1[w * HN_ + tid]; w2s[tid] = w2[w * HN_ + tid]; }
  __syncthreads();
  const long bq = (long)blockIdx.x * 256 + tid;
  float pv[D_];
#pragma unroll
  for (int d = 0; d < D_; ++d) {
    if (d < D_ - 1 - t) pv[d] = pah_init[w * D_ + d + t + 1];
    else pv[d] = pre_hist[(size_t)(d - (D_ - 1) + t) * BQ_ * W_ + bq * W_ + w];
  }
  float acc = b2[w];
#pragma unroll
  for (int hh = 0; hh < HN_; ++hh) {
    float s = b1s[hh];
#pragma unroll
    for (int d = 0; d < D_; ++d) s += pv[d] * w1s[d * HN_ + hh];
    acc += gelu_f(s) * w2s[hh];
  }
  zout[bq * W_ + w] = acc;
}

// ---------------- host orchestration ----------------
static inline void launch_gemm(hipStream_t st, const __bf16* A, const __bf16* Bm,
                               const float* bias, float* Cf, int ldcf, __bf16* Cb, int ldcb,
                               int M, int N, int K, int gelu) {
  dim3 g(N / 64, M / 64);
  gemm_bf16_kernel<<<g, 128, 0, st>>>(A, Bm, bias, Cf, ldcf, Cb, ldcb, M, N, K, gelu);
}

extern "C" void kernel_launch(void* const* d_in, const int* in_sizes, int n_in,
                              void* d_out, int out_size, void* d_ws, size_t ws_size,
                              hipStream_t stream) {
  (void)in_sizes; (void)n_in; (void)out_size; (void)ws_size;
  const int*   x        = (const int*)d_in[0];
  const float* emb      = (const float*)d_in[2];
  const float* w_kv     = (const float*)d_in[3];
  const float* w_q_sync = (const float*)d_in[4];
  const float* wq       = (const float*)d_in[5];
  const float* wk       = (const float*)d_in[6];
  const float* wv       = (const float*)d_in[7];
  const float* wo       = (const float*)d_in[8];
  const float* ws1      = (const float*)d_in[9];
  const float* bs1      = (const float*)d_in[10];
  const float* ws2      = (const float*)d_in[11];
  const float* bs2      = (const float*)d_in[12];
  const float* nlm_w1   = (const float*)d_in[13];
  const float* nlm_b1   = (const float*)d_in[14];
  const float* nlm_w2   = (const float*)d_in[15];
  const float* nlm_b2   = (const float*)d_in[16];
  const float* out_w    = (const float*)d_in[17];
  const float* out_b    = (const float*)d_in[18];
  const float* z_init   = (const float*)d_in[19];
  const float* pah_init = (const float*)d_in[20];
  const float* decay_a  = (const float*)d_in[21];
  const float* decay_o  = (const float*)d_in[22];
  const int*   idx_la   = (const int*)d_in[23];
  const int*   idx_ra   = (const int*)d_in[24];
  const int*   idx_lo   = (const int*)d_in[25];
  const int*   idx_ro   = (const int*)d_in[26];
  float* outF = (float*)d_out;

  // ---- carve workspace ----
  char* base = (char*)d_ws;
  size_t off = 0;
  auto alloc = [&](size_t bytes) -> char* {
    char* p = base + off;
    off += (bytes + 255) & ~(size_t)255;
    return p;
  };
  __bf16* wkv_bf   = (__bf16*)alloc((size_t)E_ * E_ * 2);
  __bf16* wqs_bf   = (__bf16*)alloc((size_t)SA_ * E_ * 2);
  __bf16* wq_bf    = (__bf16*)alloc((size_t)E_ * E_ * 2);
  __bf16* wk_bf    = (__bf16*)alloc((size_t)E_ * E_ * 2);
  __bf16* wv_bf    = (__bf16*)alloc((size_t)E_ * E_ * 2);
  __bf16* wo_bf    = (__bf16*)alloc((size_t)E_ * E_ * 2);
  __bf16* ws1_bf   = (__bf16*)alloc((size_t)(E_ + W_) * SH_ * 2);
  __bf16* ws2_bf   = (__bf16*)alloc((size_t)SH_ * W_ * 2);
  __bf16* outw_bf  = (__bf16*)alloc((size_t)SO_ * V_ * 2);
  __bf16* emb_bf   = (__bf16*)alloc((size_t)BQ_ * E_ * 2);
  __bf16* kv_bf    = (__bf16*)alloc((size_t)BQ_ * E_ * 2);
  float*  tmpF     = (float*)alloc((size_t)BQ_ * E_ * 4);
  __bf16* kh_bf    = (__bf16*)alloc((size_t)BQ_ * E_ * 2);
  __bf16* vt_bf    = (__bf16*)alloc((size_t)BQ_ * E_ * 2);
  __bf16* qh_bf    = (__bf16*)alloc((size_t)BQ_ * E_ * 2);
  __bf16* q_bf     = (__bf16*)alloc((size_t)BQ_ * E_ * 2);
  __bf16* attnc_bf = (__bf16*)alloc((size_t)BQ_ * E_ * 2);
  __bf16* preact_bf= (__bf16*)alloc((size_t)BQ_ * (E_ + W_) * 2);
  __bf16* h1_bf    = (__bf16*)alloc((size_t)BQ_ * SH_ * 2);
  __bf16* synca_bf = (__bf16*)alloc((size_t)BQ_ * SA_ * 2);
  __bf16* synco_bf = (__bf16*)alloc((size_t)BQ_ * SO_ * 2);
  float*  hist_f   = (float*)alloc((size_t)(T_ + 1) * BQ_ * W_ * 4);
  float*  preh_f   = (float*)alloc((size_t)T_ * BQ_ * W_ * 4);

  auto conv = [&](const float* s, __bf16* d, long n) {
    convert_bf16_kernel<<<4096, 256, 0, stream>>>(s, d, n);
  };
  // ---- weight conversions (once per call) ----
  conv(w_kv, wkv_bf, (long)E_ * E_);
  conv(w_q_sync, wqs_bf, (long)SA_ * E_);
  conv(wq, wq_bf, (long)E_ * E_);
  conv(wk, wk_bf, (long)E_ * E_);
  conv(wv, wv_bf, (long)E_ * E_);
  conv(wo, wo_bf, (long)E_ * E_);
  conv(ws1, ws1_bf, (long)(E_ + W_) * SH_);
  conv(ws2, ws2_bf, (long)SH_ * W_);
  conv(out_w, outw_bf, (long)SO_ * V_);

  // ---- pre-loop: embeddings, kv, K/V projections (loop-invariant) ----
  gather_embed_kernel<<<((long)BQ_ * E_ + 255) / 256, 256, 0, stream>>>(x, emb, emb_bf);
  launch_gemm(stream, emb_bf, wkv_bf, nullptr, nullptr, 0, kv_bf, E_, BQ_, E_, E_, 0);
  launch_gemm(stream, kv_bf, wk_bf, nullptr, tmpF, E_, nullptr, 0, BQ_, E_, E_, 0);
  rope_kernel<<<(B_ * L_ * NH_ * 32 + 255) / 256, 256, 0, stream>>>(tmpF, kh_bf, L_);
  launch_gemm(stream, kv_bf, wv_bf, nullptr, tmpF, E_, nullptr, 0, BQ_, E_, E_, 0);
  vresh_kernel<<<((long)B_ * NH_ * HD_ * L_ + 255) / 256, 256, 0, stream>>>(tmpF, vt_bf);
  hist0_kernel<<<((long)BQ_ * W_ + 255) / 256, 256, 0, stream>>>(z_init, hist_f);
  sync_kernel<<<((long)BQ_ * SA_ + 255) / 256, 256, 0, stream>>>(hist_f, 1, idx_la, idx_ra, decay_a, SA_, synca_bf);

  // ---- thought iterations ----
  for (int t = 0; t < T_; ++t) {
    // q = sync_a @ w_q_sync
    launch_gemm(stream, synca_bf, wqs_bf, nullptr, nullptr, 0, q_bf, E_, BQ_, E_, SA_, 0);
    // qh = rope(q @ wq)
    launch_gemm(stream, q_bf, wq_bf, nullptr, tmpF, E_, nullptr, 0, BQ_, E_, E_, 0);
    rope_kernel<<<(B_ * Q_ * NH_ * 32 + 255) / 256, 256, 0, stream>>>(tmpF, qh_bf, Q_);
    // attention
    attention_kernel<<<B_ * NH_ * (Q_ / 64), 128, 0, stream>>>(qh_bf, kh_bf, vt_bf, attnc_bf);
    // attn @ wo -> preact[:, 0:E], then z -> preact[:, E:E+W]
    launch_gemm(stream, attnc_bf, wo_bf, nullptr, nullptr, 0, preact_bf, E_ + W_, BQ_, E_, E_, 0);
    zcat_kernel<<<((long)BQ_ * W_ + 255) / 256, 256, 0, stream>>>(hist_f + (size_t)t * BQ_ * W_, preact_bf);
    // synapse MLP
    launch_gemm(stream, preact_bf, ws1_bf, bs1, nullptr, 0, h1_bf, SH_, BQ_, SH_, E_ + W_, 1);
    launch_gemm(stream, h1_bf, ws2_bf, bs2, preh_f + (size_t)t * BQ_ * W_, W_, nullptr, 0, BQ_, W_, SH_, 0);
    // NLM -> z (hist slab t+1)
    nlm_kernel<<<dim3(BQ_ / 256, W_), 256, 0, stream>>>(preh_f, pah_init, nlm_w1, nlm_b1, nlm_w2, nlm_b2,
                                                        hist_f + (size_t)(t + 1) * BQ_ * W_, t);
    // syncs over history (Tn = t+2)
    sync_kernel<<<((long)BQ_ * SA_ + 255) / 256, 256, 0, stream>>>(hist_f, t + 2, idx_la, idx_ra, decay_a, SA_, synca_bf);
    sync_kernel<<<((long)BQ_ * SO_ + 255) / 256, 256, 0, stream>>>(hist_f, t + 2, idx_lo, idx_ro, decay_o, SO_, synco_bf);
    // vocab projection -> output slab t
    launch_gemm(stream, synco_bf, outw_bf, out_b, outF + (size_t)t * BQ_ * V_, V_, nullptr, 0, BQ_, V_, SO_, 0);
  }
}
